// LSTM_40097814675741
// MI455X (gfx1250) — matrix operand order repaired and checked by reading, not compiled, not run on hardware
//
#include <hip/hip_runtime.h>
#include <stdint.h>

#define T_LEN   200
#define BATCH   256
#define HID     50
#define EMB     128
#define VOCAB   100000
#define GCOLS   208      // 4H = 200 padded to 13 * 16
#define NTILES  13
#define NTHREADS 416     // 13 waves of 32

typedef __bf16 v16bf __attribute__((ext_vector_type(16)));
typedef float  v8f   __attribute__((ext_vector_type(8)));
typedef unsigned int v4u __attribute__((ext_vector_type(4)));
typedef int v8i __attribute__((ext_vector_type(8)));
typedef int v4i __attribute__((ext_vector_type(4)));

#if defined(__gfx1250__) && __has_builtin(__builtin_amdgcn_tensor_load_to_lds) && \
    __has_builtin(__builtin_amdgcn_s_wait_tensorcnt)
#define USE_TDM 1
#else
#define USE_TDM 0
#endif

#define WMMA_BF16(a, b, c) \
  __builtin_amdgcn_wmma_f32_16x16x32_bf16(false, (a), false, (b), (short)0, (c), false, false)

__device__ __forceinline__ float sigmoidf_(float x) { return 1.0f / (1.0f + __expf(-x)); }

// ---- B fragment (K x N = 32 x 16 per kchunk) from row-major W[n][k] (f32 -> bf16).
// K order here (contiguous, same as A): lanes 0-15 hold K = kc*32 + 0..15, lanes 16-31 hold K = kc*32 + 16..31,
// N = lane % 16. Out-of-range n/k padded with zero.
__device__ __forceinline__ v16bf b_frag_f32(const float* __restrict__ W, int nrows, int kcols,
                                            int ntile, int kc, int lane) {
  int n = ntile * 16 + (lane & 15);
  int kbase = kc * 32 + ((lane & 16) ? 16 : 0);
  bool nok = (n < nrows);
  const float* row = W + (size_t)n * kcols;
  v16bf b;
#pragma unroll
  for (int i = 0; i < 16; ++i) {
    int k = kbase + i;
    float v = (nok && k < kcols) ? row[k] : 0.0f;
    b[i] = (__bf16)v;
  }
  return b;
}

// ---- A fragment (M x K = 16 x 32 per kchunk), bf16 source in LDS.
// K order here (contiguous, same as B): lanes 0-15: K = 0..15; lanes 16-31: K = 16..31; M = lane%16.
__device__ __forceinline__ v16bf a_frag_bf16_lds(const __bf16* src, int ldk, int lane, int kc) {
  int r = lane & 15;
  int k0 = kc * 32 + ((lane & 16) ? 16 : 0);
  const __bf16* p = src + r * ldk + k0;
  v16bf a;
#pragma unroll
  for (int i = 0; i < 16; ++i) { a[i] = p[i]; }
  return a;
}

#if USE_TDM
// Tensor Data Mover, gather mode, 32-bit row indices, 8 rows of EMB f32 each.
// D# group0: count=1, gather_index_size=1 (32b), gather_mode=1, lds_addr, global_addr, type=2.
// D# group1: data_size=4B, tensor_dim0=EMB, tensor_dim1=VOCAB, tile_dim0=EMB,
//            tile_dim1=#indices(8), tensor_dim0_stride=EMB.
// Groups 2+3 carry the 8 row indices (32-bit mode). Trailing int32x8 group zero-filled
// (6-arg toolchain variant), cpol = 0.
__device__ __forceinline__ void tdm_gather8(const float* __restrict__ table,
                                            uint32_t lds_byte_off, const int* idx8) {
  uint64_t ga = (uint64_t)(uintptr_t)table;
  v4u g0;
  g0[0] = 0x1u | (1u << 30) | (1u << 31);
  g0[1] = lds_byte_off;
  g0[2] = (uint32_t)(ga & 0xFFFFFFFFu);
  g0[3] = (uint32_t)((ga >> 32) & 0x01FFFFFFu) | (2u << 30);
  v8i g1;
  g1[0] = (int)(2u << 16);                                       // data_size = 4B
  g1[1] = (int)(((uint32_t)EMB & 0xFFFFu) << 16);                // tensor_dim0 lo16
  g1[2] = (int)(((uint32_t)EMB >> 16) | (((uint32_t)VOCAB & 0xFFFFu) << 16));
  g1[3] = (int)(((uint32_t)VOCAB >> 16) | ((uint32_t)EMB << 16)); // dim1 hi16 | tile_dim0
  g1[4] = 8;                                                      // tile_dim1 = #valid indices
  g1[5] = EMB;                                                    // tensor_dim0_stride lo32
  g1[6] = 0;
  g1[7] = 0;
  v4i g2, g3;
  g2[0] = idx8[0]; g2[1] = idx8[1]; g2[2] = idx8[2]; g2[3] = idx8[3];
  g3[0] = idx8[4]; g3[1] = idx8[5]; g3[2] = idx8[6]; g3[3] = idx8[7];
  v8i g4 = {0, 0, 0, 0, 0, 0, 0, 0};
  __builtin_amdgcn_tensor_load_to_lds(g0, g1, g2, g3, g4, 0);
}
#endif

__global__ __launch_bounds__(NTHREADS, 1)
void lstm_fused_kernel(const int* __restrict__ tokens, const float* __restrict__ embed,
                       const float* __restrict__ W_ih, const float* __restrict__ W_hh,
                       const float* __restrict__ b_ih, const float* __restrict__ b_hh,
                       const float* __restrict__ W1, const float* __restrict__ b1,
                       const float* __restrict__ W2, const float* __restrict__ b2,
                       float* __restrict__ out) {
  __shared__ __bf16 xeb[16][EMB];      // embedding rows as bf16 (WMMA-A source)
  __shared__ __bf16 hbf[16][64];       // h state in WMMA-A layout, K-padded 50->64
  __shared__ float  cst[16][HID];      // c state
  __shared__ float  gates[16][GCOLS];  // gate pre-activations (i,f,g,o)
  __shared__ float  hid[16][16];       // head hidden layer
  __shared__ float  bias_sum[GCOLS];
  __shared__ float  w2s[9][16];
  __shared__ float  b1s[16];
  __shared__ float  b2s[9];
#if USE_TDM
  __shared__ float  xe[16][EMB];       // raw f32 TDM landing buffer
#else
  __shared__ int tok_lds[16];
#endif

  const int tid  = threadIdx.x;
  const int lane = tid & 31;
  const int wid  = tid >> 5;              // 0..12 -> gate N-tile
  const int b0   = blockIdx.x * 16;       // this block's batch slice

  // ---- one-time init of LDS state ----
  for (int e = tid; e < 16 * 64; e += NTHREADS) ((__bf16*)hbf)[e] = (__bf16)0.0f;
  for (int e = tid; e < 16 * HID; e += NTHREADS) ((float*)cst)[e] = 0.0f;
  for (int e = tid; e < GCOLS; e += NTHREADS)
    bias_sum[e] = (e < 200) ? (b_ih[e] + b_hh[e]) : 0.0f;
  for (int e = tid; e < 9 * 16; e += NTHREADS) ((float*)w2s)[e] = W2[e];
  if (tid < 16) b1s[tid] = b1[tid];
  if (tid < 9)  b2s[tid] = b2[tid];

  // ---- resident bf16 B-fragments (held in VGPRs for all 200 steps) ----
  v16bf bih0 = b_frag_f32(W_ih, 200, EMB, wid, 0, lane);
  v16bf bih1 = b_frag_f32(W_ih, 200, EMB, wid, 1, lane);
  v16bf bih2 = b_frag_f32(W_ih, 200, EMB, wid, 2, lane);
  v16bf bih3 = b_frag_f32(W_ih, 200, EMB, wid, 3, lane);
  v16bf bhh0 = b_frag_f32(W_hh, 200, HID, wid, 0, lane);
  v16bf bhh1 = b_frag_f32(W_hh, 200, HID, wid, 1, lane);
  v16bf bw10 = b_frag_f32(W1, 16, HID, 0, 0, lane);
  v16bf bw11 = b_frag_f32(W1, 16, HID, 0, 1, lane);

#if USE_TDM
  const uint32_t xe_lds = (uint32_t)(uintptr_t)(&xe[0][0]);
  if (wid == 0) {  // prime the pipeline: gather embeddings for t = 0
    int idx8[8];
#pragma unroll
    for (int h = 0; h < 2; ++h) {
      for (int i = 0; i < 8; ++i) idx8[i] = tokens[b0 + h * 8 + i];
      tdm_gather8(embed, xe_lds + (uint32_t)(h * 8 * EMB * 4), idx8);
    }
  }
#endif
  __syncthreads();

  for (int t = 0; t < T_LEN; ++t) {
#if USE_TDM
    if (wid == 0) __builtin_amdgcn_s_wait_tensorcnt(0);  // xe(t) landed in LDS
    __syncthreads();
    // shared f32 -> bf16 conversion (once, instead of per-wave during A-frag builds)
    for (int e = tid; e < 16 * EMB; e += NTHREADS)
      ((__bf16*)xeb)[e] = (__bf16)(((const float*)xe)[e]);
    __syncthreads();
#else
    if (tid < 16) tok_lds[tid] = tokens[(size_t)t * BATCH + b0 + tid];
    __syncthreads();
    for (int e = tid; e < 16 * EMB; e += NTHREADS) {
      int r = e >> 7, k = e & (EMB - 1);
      xeb[r][k] = (__bf16)embed[(size_t)tok_lds[r] * EMB + k];
    }
    __syncthreads();
#endif

    // ---- gates = x * W_ih^T + h * W_hh^T + (b_ih + b_hh), one 16x16 tile per wave ----
    // Two independent accumulator chains (3 WMMAs each) to halve the RAW chain depth.
    v8f acc0 = {}, acc1 = {};
    {
      v16bf a;
      a = a_frag_bf16_lds(&xeb[0][0], EMB, lane, 0); acc0 = WMMA_BF16(a, bih0, acc0);
      a = a_frag_bf16_lds(&xeb[0][0], EMB, lane, 1); acc1 = WMMA_BF16(a, bih1, acc1);
      a = a_frag_bf16_lds(&xeb[0][0], EMB, lane, 2); acc0 = WMMA_BF16(a, bih2, acc0);
      a = a_frag_bf16_lds(&xeb[0][0], EMB, lane, 3); acc1 = WMMA_BF16(a, bih3, acc1);
      a = a_frag_bf16_lds(&hbf[0][0], 64, lane, 0);  acc0 = WMMA_BF16(a, bhh0, acc0);
      a = a_frag_bf16_lds(&hbf[0][0], 64, lane, 1);  acc1 = WMMA_BF16(a, bhh1, acc1);
    }
    const int col  = wid * 16 + (lane & 15);
    const int rofs = (lane & 16) ? 8 : 0;   // C layout: VGPR v -> M = v (+8 for hi lanes)
#pragma unroll
    for (int v = 0; v < 8; ++v) gates[v + rofs][col] = acc0[v] + acc1[v] + bias_sum[col];
    __syncthreads();

    // ---- elementwise LSTM cell (16 rows x 50) ----
    if (tid < 16 * HID) {
      int i = tid / HID, j = tid % HID;
      float ig = gates[i][j];
      float fg = gates[i][50 + j];
      float gg = gates[i][100 + j];
      float og = gates[i][150 + j];
      float cc = sigmoidf_(fg) * cst[i][j] + sigmoidf_(ig) * tanhf(gg);
      float hh = sigmoidf_(og) * tanhf(cc);
      cst[i][j] = cc;
      hbf[i][j] = (__bf16)hh;
    }
    __syncthreads();

#if USE_TDM
    if (wid == 0 && (t + 1) < T_LEN) {  // prefetch next step's embeddings during head work
      int idx8[8];
#pragma unroll
      for (int h = 0; h < 2; ++h) {
        for (int i = 0; i < 8; ++i) idx8[i] = tokens[(size_t)(t + 1) * BATCH + b0 + h * 8 + i];
        tdm_gather8(embed, xe_lds + (uint32_t)(h * 8 * EMB * 4), idx8);
      }
    }
#endif

    // ---- head layer 1: hid = relu(h @ W1^T + b1), single 16x16 tile, wave 0 ----
    if (wid == 0) {
      v8f hc = {};
      v16bf a0 = a_frag_bf16_lds(&hbf[0][0], 64, lane, 0);
      hc = WMMA_BF16(a0, bw10, hc);
      v16bf a1 = a_frag_bf16_lds(&hbf[0][0], 64, lane, 1);
      hc = WMMA_BF16(a1, bw11, hc);
      int n = lane & 15;
#pragma unroll
      for (int v = 0; v < 8; ++v) {
        float x = hc[v] + b1s[n];
        hid[v + rofs][n] = fmaxf(x, 0.0f);
      }
    }
    __syncthreads();

    // ---- head layer 2 -> out[t*B + b, 0..8] (reshape in reference is row-identity) ----
    if (tid < 16 * 9) {
      int i = tid / 9, j = tid % 9;
      float s = b2s[j];
#pragma unroll
      for (int n = 0; n < 16; ++n) s += hid[i][n] * w2s[j][n];
      out[((size_t)t * BATCH + b0 + i) * 9 + j] = s;
    }
  }
}

extern "C" void kernel_launch(void* const* d_in, const int* in_sizes, int n_in,
                              void* d_out, int out_size, void* d_ws, size_t ws_size,
                              hipStream_t stream) {
  (void)in_sizes; (void)n_in; (void)out_size; (void)d_ws; (void)ws_size;
  const int*   tokens = (const int*)d_in[0];
  const float* embed  = (const float*)d_in[1];
  const float* W_ih   = (const float*)d_in[2];
  const float* W_hh   = (const float*)d_in[3];
  const float* b_ih   = (const float*)d_in[4];
  const float* b_hh   = (const float*)d_in[5];
  const float* W1     = (const float*)d_in[6];
  const float* b1     = (const float*)d_in[7];
  const float* W2     = (const float*)d_in[8];
  const float* b2     = (const float*)d_in[9];
  float* out = (float*)d_out;

  dim3 grid(BATCH / 16);     // 16 independent batch slices (the only parallelism the scan allows)
  dim3 block(NTHREADS);      // 13 waves: one per 16-wide gate tile
  lstm_fused_kernel<<<grid, block, 0, stream>>>(tokens, embed, W_ih, W_hh, b_ih, b_hh,
                                                W1, b1, W2, b2, out);
}
